// Cxt_63883343561469
// MI455X (gfx1250) — compile-verified, matrix operand order checked by reading, not run on hardware
//
#include <hip/hip_runtime.h>
#include <hip/hip_bf16.h>
#include <math.h>

typedef __attribute__((ext_vector_type(16))) _Float16 v16h;
typedef __attribute__((ext_vector_type(8)))  _Float16 v8h;
typedef __attribute__((ext_vector_type(8)))  float    v8f;

#define BATCH 4096
#define LSEQ  264
#define LPAD  272          // 17 chunks of 16 positions
#define D0    40
#define KPAD  64           // padded channel dim for pw1 GEMM
#define HID   160          // pw hidden dim
#define NPAD2 48           // padded output dim for pw2 GEMM (3 tiles of 16)
#define WK    800          // padded K for Toeplitz GEMM (792 real + 8 zero)
#define NOUT  384

// Branchless mish: tanh(softplus(x)) = (e^2+2e)/(e^2+2e+2), e = exp(x).
// Clamp keeps n finite so the ratio -> 1 for large x (matches reference tail).
__device__ __forceinline__ float mishf(float v) {
    float e = __expf(fminf(v, 20.f));
    float n = e * (e + 2.f);
    return v * (n * __builtin_amdgcn_rcpf(n + 2.f));
}

__device__ __forceinline__ v16h pack16(v8h lo, v8h hi) {
    v16h r;
    #pragma unroll
    for (int i = 0; i < 8; ++i) { r[i] = lo[i]; r[i + 8] = hi[i]; }
    return r;
}

// ---------------------------------------------------------------------------
// Kernel 0: build padded f16 weight matrices for the pw WMMA GEMMs.
//   W1p[o][k]  : 160 x 64   (pw1_w, K zero-padded 40->64)
//   W2p[c][o]  : 48 x 160   (pw2_w, N zero-padded 40->48)
// ---------------------------------------------------------------------------
__global__ void build_w(const float* __restrict__ pw1_w,
                        const float* __restrict__ pw2_w,
                        _Float16* __restrict__ W1p, _Float16* __restrict__ W2p)
{
    int i = blockIdx.x * blockDim.x + threadIdx.x;
    if (i < HID * KPAD) {
        int o = i >> 6, k = i & 63;
        W1p[i] = (_Float16)((k < D0) ? pw1_w[o * D0 + k] : 0.f);
    }
    if (i < NPAD2 * HID) {
        int c = i / HID, o = i - c * HID;
        W2p[i] = (_Float16)((c < D0) ? pw2_w[c * HID + o] : 0.f);
    }
}

// ---------------------------------------------------------------------------
// Kernel 1: fused per-row pipeline. One workgroup (8 waves) per batch row.
// LN0 -> conv1+mish -> dw conv -> channel-LN -> [WMMA pw1 -> mish -> WMMA pw2
// + residual] per 16-position chunk -> 1x1 stack -> f16 activation row (800).
// ---------------------------------------------------------------------------
__global__ __launch_bounds__(256) void fused_pre(
    const float* __restrict__ x,
    const float* __restrict__ ln0_w, const float* __restrict__ ln0_b,
    const float* __restrict__ conv1_w, const float* __restrict__ conv1_b,
    const float* __restrict__ dw_w,   const float* __restrict__ dw_b,
    const float* __restrict__ bn_w,   const float* __restrict__ bn_b,
    const _Float16* __restrict__ W1p, const float* __restrict__ pw1_b,
    const _Float16* __restrict__ W2p, const float* __restrict__ pw2_b,
    const float* __restrict__ gamma_p,
    const float* __restrict__ c0_w, const float* __restrict__ c0_b,
    const float* __restrict__ c1_w, const float* __restrict__ c1_b,
    const float* __restrict__ c2_w, const float* __restrict__ c2_b,
    const float* __restrict__ c3_w, const float* __restrict__ c3_b,
    _Float16* __restrict__ act)
{
    __shared__ float s_xln[LSEQ];
    __shared__ __align__(16) _Float16 s_h1t[LPAD * D0];   // conv1+mish, [l][c]
    __shared__ __align__(16) _Float16 s_hnt[LPAD * KPAD]; // hn / x2, [l][k], K-padded
    __shared__ __align__(16) _Float16 s_g[16 * HID];      // mish(pw1) chunk, [m][o]
    __shared__ float s_red[2];

    const int tid = threadIdx.x;
    const int b   = blockIdx.x;

    if (tid < 2) s_red[tid] = 0.f;
    for (int i = tid; i < LPAD * KPAD; i += 256) s_hnt[i] = (_Float16)0.f;
    for (int i = tid; i < LPAD * D0;   i += 256) s_h1t[i] = (_Float16)0.f;
    for (int i = tid; i < LSEQ; i += 256) s_xln[i] = x[b * LSEQ + i];
    __syncthreads();

    // ---- LayerNorm over the full 264-row (C=1), eps 1e-5 ----
    float ps = 0.f, pq = 0.f;
    for (int i = tid; i < LSEQ; i += 256) { float v = s_xln[i]; ps += v; pq += v * v; }
    atomicAdd(&s_red[0], ps);
    atomicAdd(&s_red[1], pq);
    __syncthreads();
    const float mu0   = s_red[0] * (1.f / LSEQ);
    const float var0  = s_red[1] * (1.f / LSEQ) - mu0 * mu0;
    const float rstd0 = rsqrtf(var0 + 1e-5f);
    for (int i = tid; i < LSEQ; i += 256)
        s_xln[i] = (s_xln[i] - mu0) * rstd0 * ln0_w[i] + ln0_b[i];
    __syncthreads();

    // ---- conv1 (1->40, k=15, pad=7) + mish -> s_h1t[l][c] ----
    for (int idx = tid; idx < D0 * LSEQ; idx += 256) {
        int c = idx / LSEQ, l = idx - c * LSEQ;
        float s = conv1_b[c];
        #pragma unroll
        for (int k = 0; k < 15; ++k) {
            int p = l + k - 7;
            if (p >= 0 && p < LSEQ) s += s_xln[p] * conv1_w[c * 15 + k];
        }
        s_h1t[l * D0 + c] = (_Float16)mishf(s);
    }
    __syncthreads();

    // ---- depthwise conv (k=15, pad=7) -> s_hnt[l][c] ----
    for (int idx = tid; idx < D0 * LSEQ; idx += 256) {
        int c = idx / LSEQ, l = idx - c * LSEQ;
        float s = dw_b[c];
        #pragma unroll
        for (int k = 0; k < 15; ++k) {
            int p = l + k - 7;
            if (p >= 0 && p < LSEQ) s += (float)s_h1t[p * D0 + c] * dw_w[c * 15 + k];
        }
        s_hnt[l * KPAD + c] = (_Float16)s;
    }
    __syncthreads();

    // ---- channel LayerNorm (40 ch) + affine, eps 1e-6, in place ----
    for (int l = tid; l < LSEQ; l += 256) {
        float s = 0.f, q = 0.f;
        #pragma unroll 8
        for (int c = 0; c < D0; ++c) { float v = (float)s_hnt[l * KPAD + c]; s += v; q += v * v; }
        float mu   = s * (1.f / D0);
        float var  = q * (1.f / D0) - mu * mu;
        float rstd = rsqrtf(var + 1e-6f);
        #pragma unroll 8
        for (int c = 0; c < D0; ++c) {
            float v = ((float)s_hnt[l * KPAD + c] - mu) * rstd;
            s_hnt[l * KPAD + c] = (_Float16)(bn_w[c] * v + bn_b[c]);
        }
    }

    // ---- WMMA pw1 (16x64 . 64x160) -> mish -> WMMA pw2 (16x160 . 160x48)
    //      per 16-position chunk; residual fused into the C-fragment epilogue.
    const int wave   = tid >> 5;
    const int lane   = tid & 31;
    const int half16 = lane >> 4;            // 0 | 1
    const int mn     = lane & 15;            // row (A) / column (B,C) in tile
    const int kb     = half16 * 8;           // K sub-group offset per ISA layout

    for (int ck = 0; ck < LPAD / 16; ++ck) {
        const int l0 = ck * 16;
        __syncthreads();                     // s_g reuse + hn ready

        // GEMM1: hn_chunk x W1p  (10 N-tiles over 8 waves)
        for (int t = wave; t < HID / 16; t += 8) {
            v8f acc;
            #pragma unroll
            for (int e = 0; e < 8; ++e) acc[e] = 0.f;
            #pragma unroll
            for (int ks = 0; ks < KPAD / 32; ++ks) {
                const int k0 = ks * 32;
                const _Float16* ap = &s_hnt[(l0 + mn) * KPAD + k0 + kb];
                v16h af = pack16(*(const v8h*)ap, *(const v8h*)(ap + 16));
                const _Float16* bp = W1p + (t * 16 + mn) * KPAD + k0 + kb;
                v16h bf = pack16(*(const v8h*)bp, *(const v8h*)(bp + 16));
                acc = __builtin_amdgcn_wmma_f32_16x16x32_f16(
                    false, af, false, bf, (short)0, acc, false, false);
            }
            const float bo = pw1_b[t * 16 + mn];
            #pragma unroll
            for (int e = 0; e < 8; ++e)
                s_g[(e + 8 * half16) * HID + t * 16 + mn] = (_Float16)mishf(acc[e] + bo);
        }
        __syncthreads();

        // GEMM2: g_chunk x W2p  (3 N-tiles on waves 0..2), fused epilogue
        if (wave < 3) {
            v8f acc;
            #pragma unroll
            for (int e = 0; e < 8; ++e) acc[e] = 0.f;
            #pragma unroll
            for (int ks = 0; ks < HID / 32; ++ks) {
                const int k0 = ks * 32;
                const _Float16* ap = &s_g[mn * HID + k0 + kb];
                v16h af = pack16(*(const v8h*)ap, *(const v8h*)(ap + 16));
                const _Float16* bp = W2p + (wave * 16 + mn) * HID + k0 + kb;
                v16h bf = pack16(*(const v8h*)bp, *(const v8h*)(bp + 16));
                acc = __builtin_amdgcn_wmma_f32_16x16x32_f16(
                    false, af, false, bf, (short)0, acc, false, false);
            }
            const int c2 = wave * 16 + mn;
            if (c2 < D0) {
                const float gg = gamma_p[c2];
                const float bb = pw2_b[c2];
                #pragma unroll
                for (int e = 0; e < 8; ++e) {
                    const int l = l0 + e + 8 * half16;
                    float xv = (float)s_h1t[l * D0 + c2] + gg * (acc[e] + bb);
                    s_hnt[l * KPAD + c2] = (_Float16)xv;  // overwrite consumed hn
                }
            }
        }
    }
    __syncthreads();

    // ---- 1x1 stack 40->30->30->10->3 with mish, write f16 activation row ----
    for (int l = tid; l < LSEQ; l += 256) {
        float x2[D0];
        #pragma unroll 8
        for (int c = 0; c < D0; ++c) x2[c] = (float)s_hnt[l * KPAD + c];

        float v0[30];
        for (int o = 0; o < 30; ++o) {
            float t = c0_b[o];
            #pragma unroll 8
            for (int c = 0; c < D0; ++c) t += c0_w[o * D0 + c] * x2[c];
            v0[o] = mishf(t);
        }
        float v1[30];
        for (int o = 0; o < 30; ++o) {
            float t = c1_b[o];
            #pragma unroll 6
            for (int c = 0; c < 30; ++c) t += c1_w[o * 30 + c] * v0[c];
            v1[o] = mishf(t);
        }
        float v2[10];
        for (int o = 0; o < 10; ++o) {
            float t = c2_b[o];
            #pragma unroll 6
            for (int c = 0; c < 30; ++c) t += c2_w[o * 30 + c] * v1[c];
            v2[o] = mishf(t);
        }
        #pragma unroll
        for (int o = 0; o < 3; ++o) {
            float t = c3_b[o];
            #pragma unroll
            for (int c = 0; c < 10; ++c) t += c3_w[o * 10 + c] * v2[c];
            act[(size_t)b * WK + o * LSEQ + l] = (_Float16)mishf(t);
        }
    }
    if (tid < 8) act[(size_t)b * WK + 792 + tid] = (_Float16)0.f;
}

// ---------------------------------------------------------------------------
// Kernel 2: dense f16 Toeplitz operator, K-major: Tt[n][k] = toep_w[k-n+383].
// ---------------------------------------------------------------------------
__global__ void build_T(const float* __restrict__ toep_w, _Float16* __restrict__ Tt)
{
    int idx = blockIdx.x * blockDim.x + threadIdx.x;
    if (idx >= NOUT * WK) return;
    int n = idx / WK, k = idx - n * WK;
    float v = 0.f;
    if (k < 792) v = toep_w[k - n + 383];   // index always in [0,1175) here
    Tt[idx] = (_Float16)v;
}

// ---------------------------------------------------------------------------
// Kernel 3: WMMA f16 GEMM (4096x800)·Tt + fused row softmax.
// ---------------------------------------------------------------------------
__global__ __launch_bounds__(256) void toep_gemm_softmax(
    const _Float16* __restrict__ act, const _Float16* __restrict__ Tt,
    float* __restrict__ out)
{
    __shared__ __align__(16) _Float16 sA[16 * WK];
    __shared__ float sL[16 * NOUT];
    __shared__ float sRed[16 * 16];

    const int tid = threadIdx.x;
    const int m0  = blockIdx.x * 16;

    {
        const uint4* g = (const uint4*)(act + (size_t)m0 * WK);
        uint4*       s = (uint4*)sA;
        for (int i = tid; i < 16 * WK / 8; i += 256) s[i] = g[i];
    }
    __syncthreads();

    const int wave   = tid >> 5;
    const int lane   = tid & 31;
    const int half16 = lane >> 4;
    const int mn     = lane & 15;
    const int kb     = half16 * 8;

    v8f acc[3];
    #pragma unroll
    for (int j = 0; j < 3; ++j)
        #pragma unroll
        for (int e = 0; e < 8; ++e) acc[j][e] = 0.f;

    for (int ks = 0; ks < WK / 32; ++ks) {
        const int k0 = ks * 32;
        const _Float16* ap = &sA[mn * WK + k0 + kb];
        v16h af = pack16(*(const v8h*)ap, *(const v8h*)(ap + 16));
        #pragma unroll
        for (int j = 0; j < 3; ++j) {
            const int n = wave * 48 + j * 16 + mn;
            const _Float16* bp = Tt + (size_t)n * WK + k0 + kb;
            v16h bf = pack16(*(const v8h*)bp, *(const v8h*)(bp + 16));
            acc[j] = __builtin_amdgcn_wmma_f32_16x16x32_f16(
                false, af, false, bf, (short)0, acc[j], false, false);
        }
    }

    #pragma unroll
    for (int j = 0; j < 3; ++j) {
        const int n = wave * 48 + j * 16 + mn;
        #pragma unroll
        for (int e = 0; e < 8; ++e)
            sL[(e + 8 * half16) * NOUT + n] = acc[j][e];
    }
    __syncthreads();

    const int r  = tid >> 4;
    const int ch = tid & 15;            // 24 columns per thread
    float mx = -3.402823466e38f;
    for (int c = 0; c < 24; ++c) mx = fmaxf(mx, sL[r * NOUT + ch * 24 + c]);
    sRed[r * 16 + ch] = mx;
    __syncthreads();
    float rmx = -3.402823466e38f;
    #pragma unroll
    for (int i = 0; i < 16; ++i) rmx = fmaxf(rmx, sRed[r * 16 + i]);
    float psum = 0.f;
    for (int c = 0; c < 24; ++c) {
        float e = __expf(sL[r * NOUT + ch * 24 + c] - rmx);
        sL[r * NOUT + ch * 24 + c] = e;
        psum += e;
    }
    __syncthreads();
    sRed[r * 16 + ch] = psum;
    __syncthreads();
    float rs = 0.f;
    #pragma unroll
    for (int i = 0; i < 16; ++i) rs += sRed[r * 16 + i];
    const float inv = 1.f / rs;
    for (int c = 0; c < 24; ++c)
        out[(size_t)(m0 + r) * NOUT + ch * 24 + c] = sL[r * NOUT + ch * 24 + c] * inv;
}

// ---------------------------------------------------------------------------
extern "C" void kernel_launch(void* const* d_in, const int* in_sizes, int n_in,
                              void* d_out, int out_size, void* d_ws, size_t ws_size,
                              hipStream_t stream)
{
    (void)in_sizes; (void)n_in; (void)out_size; (void)ws_size;

    const float* x       = (const float*)d_in[0];
    const float* ln0_w   = (const float*)d_in[1];
    const float* ln0_b   = (const float*)d_in[2];
    const float* conv1_w = (const float*)d_in[3];
    const float* conv1_b = (const float*)d_in[4];
    const float* dw_w    = (const float*)d_in[5];
    const float* dw_b    = (const float*)d_in[6];
    const float* bn_w    = (const float*)d_in[7];
    const float* bn_b    = (const float*)d_in[8];
    const float* pw1_w   = (const float*)d_in[9];
    const float* pw1_b   = (const float*)d_in[10];
    const float* pw2_w   = (const float*)d_in[11];
    const float* pw2_b   = (const float*)d_in[12];
    const float* gamma_p = (const float*)d_in[13];
    const float* c0_w    = (const float*)d_in[14];
    const float* c0_b    = (const float*)d_in[15];
    const float* c1_w    = (const float*)d_in[16];
    const float* c1_b    = (const float*)d_in[17];
    const float* c2_w    = (const float*)d_in[18];
    const float* c2_b    = (const float*)d_in[19];
    const float* c3_w    = (const float*)d_in[20];
    const float* c3_b    = (const float*)d_in[21];
    const float* toep_w  = (const float*)d_in[22];

    char* ws = (char*)d_ws;
    _Float16* act = (_Float16*)ws;                            ws += (size_t)BATCH * WK * 2;
    _Float16* Tt  = (_Float16*)ws;                            ws += (size_t)NOUT * WK * 2;
    _Float16* W1p = (_Float16*)ws;                            ws += (size_t)HID * KPAD * 2;
    _Float16* W2p = (_Float16*)ws;
    float*    out = (float*)d_out;

    build_w<<<(HID * KPAD + 255) / 256, 256, 0, stream>>>(pw1_w, pw2_w, W1p, W2p);
    build_T<<<(NOUT * WK + 255) / 256, 256, 0, stream>>>(toep_w, Tt);

    fused_pre<<<BATCH, 256, 0, stream>>>(
        x, ln0_w, ln0_b, conv1_w, conv1_b, dw_w, dw_b, bn_w, bn_b,
        W1p, pw1_b, W2p, pw2_b, gamma_p,
        c0_w, c0_b, c1_w, c1_b, c2_w, c2_b, c3_w, c3_b, act);

    toep_gemm_softmax<<<BATCH / 16, 256, 0, stream>>>(act, Tt, out);
}
